// HFB_81484119540017
// MI455X (gfx1250) — matrix operand order repaired and checked by reading, not compiled, not run on hardware
//
#include <hip/hip_runtime.h>
#include <math.h>

// ---- problem constants (match reference) ----
#define BB   4
#define CC   128
#define HH   64
#define WW   64
#define NN   4096        // H*W
#define CHH  340
#define MB   64          // attention m-block (K columns per inner step)

typedef __attribute__((ext_vector_type(16))) _Float16 v16h;
typedef __attribute__((ext_vector_type(8)))  float    v8f;

// ---------------------------------------------------------------------------
// Channel LayerNorm over NCHW channel dim: one 128-thread block per pixel.
// ---------------------------------------------------------------------------
__global__ __launch_bounds__(128)
void ln_c_kernel(const float* __restrict__ src, const float* __restrict__ gam,
                 const float* __restrict__ bet, float* __restrict__ dst)
{
    __shared__ float red[128];
    const int pix = blockIdx.x;          // b*NN + n
    const int c   = threadIdx.x;         // channel
    const int b   = pix / NN;
    const int n   = pix % NN;
    const size_t base = ((size_t)b * CC + c) * NN + n;
    float v = src[base];
    red[c] = v;
    __syncthreads();
    for (int off = 64; off > 0; off >>= 1) {
        if (c < off) red[c] += red[c + off];
        __syncthreads();
    }
    const float mu = red[0] * (1.0f / CC);
    __syncthreads();
    const float d = v - mu;
    red[c] = d * d;
    __syncthreads();
    for (int off = 64; off > 0; off >>= 1) {
        if (c < off) red[c] += red[c + off];
        __syncthreads();
    }
    const float var = red[0] * (1.0f / CC);
    dst[base] = d * rsqrtf(var + 1e-5f) * gam[c] + bet[c];
}

// ---------------------------------------------------------------------------
// 1x1 conv + folded BN (+ optional SiLU / SiLU->GELU) + optional residual.
// idx layout: ((b*Cout + co)*NN + n); threads along n -> coalesced src reads.
// ---------------------------------------------------------------------------
__global__ __launch_bounds__(256)
void pw_conv_kernel(const float* __restrict__ src, const float* __restrict__ Wmat,
                    const float* __restrict__ scl, const float* __restrict__ bia,
                    const float* __restrict__ residual, float* __restrict__ dst,
                    int Cin, int Cout, int act)
{
    const int idx = blockIdx.x * blockDim.x + threadIdx.x;
    const int n  = idx % NN;
    const int co = (idx / NN) % Cout;
    const int b  = idx / (NN * Cout);
    const float* sp = src + (size_t)b * Cin * NN + n;
    const float* wp = Wmat + (size_t)co * Cin;
    float acc = 0.0f;
    for (int ci = 0; ci < Cin; ++ci)
        acc = fmaf(wp[ci], sp[(size_t)ci * NN], acc);
    float v = acc * scl[co] + bia[co];
    if (act >= 1) v = v / (1.0f + __expf(-v));                            // SiLU
    if (act == 2) v = 0.5f * v * (1.0f + erff(v * 0.70710678118654752f)); // erf-GELU
    if (residual) v += residual[idx];
    dst[idx] = v;
}

// ---------------------------------------------------------------------------
// Depthwise 3x3 conv (SAME, zero pad) + folded BN + SiLU, emits f16.
// ---------------------------------------------------------------------------
__global__ __launch_bounds__(256)
void dw_conv_kernel(const float* __restrict__ src, const float* __restrict__ wt,
                    const float* __restrict__ scl, const float* __restrict__ bia,
                    _Float16* __restrict__ dst)
{
    const int idx = blockIdx.x * blockDim.x + threadIdx.x;   // b*C*H*W
    const int x = idx % WW;
    const int y = (idx / WW) % HH;
    const int c = (idx / (HH * WW)) % CC;
    const int b = idx / (CC * HH * WW);
    const float* im = src + ((size_t)b * CC + c) * (HH * WW);
    const float* w9 = wt + c * 9;
    float acc = 0.0f;
#pragma unroll
    for (int ky = 0; ky < 3; ++ky) {
        const int yy = y + ky - 1;
        if (yy < 0 || yy >= HH) continue;
#pragma unroll
        for (int kx = 0; kx < 3; ++kx) {
            const int xx = x + kx - 1;
            if (xx < 0 || xx >= WW) continue;
            acc = fmaf(im[yy * WW + xx], w9[ky * 3 + kx], acc);
        }
    }
    float v = acc * scl[c] + bia[c];
    v = v / (1.0f + __expf(-v));   // SiLU
    dst[idx] = (_Float16)v;
}

// ---------------------------------------------------------------------------
// Flash attention on WMMA.
//   out[b,c,n] = x_s[b,c,n] + sum_m softmax_m(q^T k / alpha)[n,m] * k[b,c,m]
// Workgroup = 128 threads = 4 waves; wave w owns 16 query rows (n-tile).
// Per 64-column K block:
//   S (16x64) = q^T(16x128) x k(128x64)      -> 16x v_wmma_f32_16x16x32_f16
//   online softmax (16-lane shuffle reductions amortized over 64 cols)
//   O (16x128) += P(16x64) x k^T(64x128)     -> 16x v_wmma_f32_16x16x32_f16
// K block staged in LDS in both [c][m] and [m][c] so every B fragment is a
// contiguous 32B ds read; shared across the 4 waves. Next K block is
// prefetched (global_prefetch_b8) while the current block computes.
// ---------------------------------------------------------------------------
__global__ __launch_bounds__(128)
void attn_kernel(const _Float16* __restrict__ Q, const _Float16* __restrict__ K,
                 const float* __restrict__ x_s, const float* __restrict__ alpha_p,
                 float* __restrict__ out)
{
    __shared__ alignas(32) _Float16 lds_k [CC * MB];      // [c][m]  16 KB
    __shared__ alignas(32) _Float16 lds_kT[MB * CC];      // [m][c]  16 KB
    __shared__ alignas(32) _Float16 lds_p [4 * 16 * MB];  // per-wave P stage 8 KB

    const int tid  = threadIdx.x;
    const int wave = tid >> 5;
    const int lane = tid & 31;
    const int g    = lane >> 4;     // lane group (ISA: splits K for A / B frags)
    const int l15  = lane & 15;

    const int blocks_per_batch = NN / 64;                 // 64 rows per WG
    const int b  = blockIdx.x / blocks_per_batch;
    const int n0 = (blockIdx.x % blocks_per_batch) * 64 + wave * 16;

    const float inv_alpha = 1.0f / alpha_p[0];
    const _Float16* Qb = Q + (size_t)b * CC * NN;
    const _Float16* Kb = K + (size_t)b * CC * NN;

    // ---- preload A fragments of q^T (16 rows x 128 K) : 4 K-steps ----
    // A frag, this code's k order (B matches): lane row M = lane%16; element e -> K:
    //   g*16+e
    v16h aq[4];
    const int nrow = n0 + l15;
    for (int ks = 0; ks < 4; ++ks) {
        v16h a;
#pragma unroll
        for (int e = 0; e < 16; ++e) {
            const int kk = g * 16 + e;
            a[e] = Qb[(size_t)(ks * 32 + kk) * NN + nrow];
        }
        aq[ks] = a;
    }

    v8f acc[8];
#pragma unroll
    for (int t = 0; t < 8; ++t) acc[t] = v8f{};
    float run_max[8], run_sum[8];
#pragma unroll
    for (int r = 0; r < 8; ++r) { run_max[r] = -INFINITY; run_sum[r] = 0.0f; }

    for (int m0 = 0; m0 < NN; m0 += MB) {
        __syncthreads();
        {   // cooperative K-block load: thread t = channel c, 64 cols
            const int c = tid;
            const _Float16* src = Kb + (size_t)c * NN + m0;
            v16h t4[4];
#pragma unroll
            for (int i = 0; i < 4; ++i) t4[i] = *(const v16h*)(src + i * 16);
#pragma unroll
            for (int i = 0; i < 4; ++i) *(v16h*)&lds_k[c * MB + i * 16] = t4[i];
#pragma unroll
            for (int i = 0; i < 4; ++i)
#pragma unroll
                for (int m = 0; m < 16; ++m)
                    lds_kT[(i * 16 + m) * CC + c] = t4[i][m];
            if (m0 + MB < NN)
                __builtin_prefetch(src + MB, 0, 3);   // global_prefetch_b8
        }
        __syncthreads();

        // ---- S = q^T k : four 16x16 m-tiles, K=128 in 4 steps each ----
        v8f s[4];
#pragma unroll
        for (int mt = 0; mt < 4; ++mt) {
            v8f sa = v8f{};
#pragma unroll
            for (int ks = 0; ks < 4; ++ks) {
                // B frag from lds_kT[m][c]: lane col m = mt*16+l15,
                // elements e -> K = ks*32 + g*16 + e  (contiguous 32B)
                const v16h bf = *(const v16h*)&lds_kT[(mt * 16 + l15) * CC + ks * 32 + g * 16];
                sa = __builtin_amdgcn_wmma_f32_16x16x32_f16(
                        false, aq[ks], false, bf, (short)0, sa, false, false);
            }
            s[mt] = sa;
        }

        // ---- online softmax (row = r + 8*g across lane groups) ----
        _Float16* pl = &lds_p[wave * 16 * MB];
#pragma unroll
        for (int r = 0; r < 8; ++r) {
            const float v0 = s[0][r] * inv_alpha;
            const float v1 = s[1][r] * inv_alpha;
            const float v2 = s[2][r] * inv_alpha;
            const float v3 = s[3][r] * inv_alpha;
            float mx = fmaxf(fmaxf(v0, v1), fmaxf(v2, v3));
#pragma unroll
            for (int off = 1; off < 16; off <<= 1)
                mx = fmaxf(mx, __shfl_xor(mx, off));
            const float nm   = fmaxf(run_max[r], mx);
            const float corr = __expf(run_max[r] - nm);
            const float e0 = __expf(v0 - nm);
            const float e1 = __expf(v1 - nm);
            const float e2 = __expf(v2 - nm);
            const float e3 = __expf(v3 - nm);
            float rs = (e0 + e1) + (e2 + e3);
#pragma unroll
            for (int off = 1; off < 16; off <<= 1)
                rs += __shfl_xor(rs, off);
            run_sum[r] = run_sum[r] * corr + rs;
            run_max[r] = nm;
            // stage P directly to per-wave LDS (D layout -> [row][col])
            const int row = r + 8 * g;
            pl[row * MB +      l15] = (_Float16)e0;
            pl[row * MB + 16 + l15] = (_Float16)e1;
            pl[row * MB + 32 + l15] = (_Float16)e2;
            pl[row * MB + 48 + l15] = (_Float16)e3;
#pragma unroll
            for (int t = 0; t < 8; ++t) acc[t][r] *= corr;   // rescale O
        }

        // ---- read P back as two A fragments (16x32 each) ----
        v16h ap[2];
#pragma unroll
        for (int pk = 0; pk < 2; ++pk) {
            v16h a;
#pragma unroll
            for (int e = 0; e < 16; ++e) {
                const int kk = g * 16 + e;
                a[e] = pl[l15 * MB + pk * 32 + kk];
            }
            ap[pk] = a;
        }

        // ---- O += P * k^T : 8 c-tiles x 2 K-steps ----
#pragma unroll
        for (int ct = 0; ct < 8; ++ct) {
#pragma unroll
            for (int pk = 0; pk < 2; ++pk) {
                // B frag from lds_k[c][m]: lane col c = ct*16+l15,
                // elements e -> K(m) = pk*32 + g*16 + e  (contiguous 32B)
                const v16h bk = *(const v16h*)&lds_k[(ct * 16 + l15) * MB + pk * 32 + g * 16];
                acc[ct] = __builtin_amdgcn_wmma_f32_16x16x32_f16(
                            false, ap[pk], false, bk, (short)0, acc[ct], false, false);
            }
        }
    }

    // ---- finalize: divide by softmax denom, add residual x_s, store ----
    float* ob = out + (size_t)b * CC * NN;
    const float* xb = x_s + (size_t)b * CC * NN;
#pragma unroll
    for (int ct = 0; ct < 8; ++ct) {
        const int c = ct * 16 + l15;
#pragma unroll
        for (int r = 0; r < 8; ++r) {
            const int n = n0 + r + 8 * g;
            const size_t o = (size_t)c * NN + n;
            ob[o] = xb[o] + acc[ct][r] / run_sum[r];
        }
    }
}

// ---------------------------------------------------------------------------
extern "C" void kernel_launch(void* const* d_in, const int* in_sizes, int n_in,
                              void* d_out, int out_size, void* d_ws, size_t ws_size,
                              hipStream_t stream)
{
    (void)in_sizes; (void)n_in; (void)out_size; (void)ws_size;

    const float* x_h   = (const float*)d_in[0];
    const float* x_s   = (const float*)d_in[1];
    const float* alpha = (const float*)d_in[2];
    const float* q_w1  = (const float*)d_in[3];
    const float* q_s1  = (const float*)d_in[4];
    const float* q_b1  = (const float*)d_in[5];
    const float* q_w2  = (const float*)d_in[6];
    const float* q_s2  = (const float*)d_in[7];
    const float* q_b2  = (const float*)d_in[8];
    const float* kv_w1 = (const float*)d_in[9];
    const float* kv_s1 = (const float*)d_in[10];
    const float* kv_b1 = (const float*)d_in[11];
    const float* kv_w2 = (const float*)d_in[12];
    const float* kv_s2 = (const float*)d_in[13];
    const float* kv_b2 = (const float*)d_in[14];
    const float* ln_g  = (const float*)d_in[15];
    const float* ln_b  = (const float*)d_in[16];
    const float* m_w1  = (const float*)d_in[17];
    const float* m_s1  = (const float*)d_in[18];
    const float* m_b1  = (const float*)d_in[19];
    const float* m_w2  = (const float*)d_in[20];
    const float* m_s2  = (const float*)d_in[21];
    const float* m_b2  = (const float*)d_in[22];

    const size_t elems = (size_t)BB * CC * NN;          // 2,097,152

    // workspace partition (all offsets 64B-aligned by construction)
    float*     xsn = (float*)d_ws;                      //  8 MB  LN output (reused)
    float*     t1  = xsn + elems;                       //  8 MB  conv1 temps
    float*     xf  = t1 + elems;                        //  8 MB  x_fused
    _Float16*  qh  = (_Float16*)(xf + elems);           //  4 MB  q  (f16)
    _Float16*  kh  = qh + elems;                        //  4 MB  k  (f16)
    float*     mh  = (float*)(kh + elems);              // 22 MB  MLP hidden [B,CH,N]

    const int ln_grid = BB * NN;                        // 16384 blocks x 128
    const int pwC_grid  = (int)(elems / 256);           // Cout = 128
    const int pwCH_grid = (BB * CHH * NN) / 256;        // Cout = 340
    const int dw_grid   = (int)(elems / 256);
    const int at_grid   = BB * (NN / 64);               // 256 blocks x 128

    // 1) LayerNorm(x_s) -> xsn
    ln_c_kernel<<<ln_grid, 128, 0, stream>>>(x_s, ln_g, ln_b, xsn);
    // 2) q path: 1x1 conv-bn-silu -> dw 3x3 conv-bn-silu (f16)
    pw_conv_kernel<<<pwC_grid, 256, 0, stream>>>(x_h, q_w1, q_s1, q_b1, nullptr, t1, CC, CC, 1);
    dw_conv_kernel<<<dw_grid, 256, 0, stream>>>(t1, q_w2, q_s2, q_b2, qh);
    // 3) kv path on LN'd x_s
    pw_conv_kernel<<<pwC_grid, 256, 0, stream>>>(xsn, kv_w1, kv_s1, kv_b1, nullptr, t1, CC, CC, 1);
    dw_conv_kernel<<<dw_grid, 256, 0, stream>>>(t1, kv_w2, kv_s2, kv_b2, kh);
    // 4) flash attention + residual -> x_fused
    attn_kernel<<<at_grid, 128, 0, stream>>>(qh, kh, x_s, alpha, xf);
    // 5) LayerNorm(x_fused) -> xsn (reuse)
    ln_c_kernel<<<ln_grid, 128, 0, stream>>>(xf, ln_g, ln_b, xsn);
    // 6) MLP up: 1x1 conv-bn-silu then erf-GELU -> mh
    pw_conv_kernel<<<pwCH_grid, 256, 0, stream>>>(xsn, m_w1, m_s1, m_b1, nullptr, mh, CC, CHH, 2);
    // 7) MLP down: 1x1 conv-bn (no act) + residual x_fused -> d_out
    pw_conv_kernel<<<pwC_grid, 256, 0, stream>>>(mh, m_w2, m_s2, m_b2, xf, (float*)d_out, CHH, CC, 0);
}
